// Attention_12790412607562
// MI455X (gfx1250) — matrix operand order repaired and checked by reading, not compiled, not run on hardware
//
#include <hip/hip_runtime.h>

// MI455X / gfx1250, wave32. Attention fwd: context + full attention matrix.
// B=4 H=16 S=2048 D=64, fp32 in/out, f16 WMMA (f32 accumulate) internally.

typedef _Float16 v16h __attribute__((ext_vector_type(16)));
typedef _Float16 v8h  __attribute__((ext_vector_type(8)));
typedef float    v8f  __attribute__((ext_vector_type(8)));
typedef float    v4f  __attribute__((ext_vector_type(4)));

#define B_ 4
#define H_ 16
#define S_ 2048
#define D_ 64
#define NTILES (S_ / 16)          // 128 key tiles per row-block
#define NCHUNK (S_ / 32)          // 64 32-key chunks
#define WAVES 8                   // waves per block
#define PSTRIDE 40                // halfs per row in LDS tiles (80B, 16B aligned)

__device__ __forceinline__ v8f wmma_f16(v16h a, v16h b, v8f c) {
  return __builtin_amdgcn_wmma_f32_16x16x32_f16(
      /*neg_a=*/false, a, /*neg_b=*/false, b,
      /*c_mod=*/(short)0, c, /*reuse_a=*/false, /*reuse_b=*/false);
}

// A-fragment (Q, 16x32 f16): lane holds row=ln, K = 16hi+0..15 (contiguous)
__device__ __forceinline__ v16h load_qa(const float* __restrict__ qrow, int c, int hi) {
  const int base = c * 32 + 16 * hi;
  v4f a0 = *(const v4f*)(qrow + base);
  v4f a1 = *(const v4f*)(qrow + base + 4);
  v4f a2 = *(const v4f*)(qrow + base + 8);
  v4f a3 = *(const v4f*)(qrow + base + 12);
  v16h r;
#pragma unroll
  for (int i = 0; i < 4; ++i) {
    r[i]      = (_Float16)a0[i];
    r[4 + i]  = (_Float16)a1[i];
    r[8 + i]  = (_Float16)a2[i];
    r[12 + i] = (_Float16)a3[i];
  }
  return r;
}

// B-fragment (K^T, 32x16 f16): lane holds col n=ln, 16 contiguous dims
__device__ __forceinline__ v16h load_kb(const float* __restrict__ p) {
  v4f x0 = *(const v4f*)(p + 0);
  v4f x1 = *(const v4f*)(p + 4);
  v4f x2 = *(const v4f*)(p + 8);
  v4f x3 = *(const v4f*)(p + 12);
  v16h r;
#pragma unroll
  for (int i = 0; i < 4; ++i) {
    r[i]      = (_Float16)x0[i];
    r[4 + i]  = (_Float16)x1[i];
    r[8 + i]  = (_Float16)x2[i];
    r[12 + i] = (_Float16)x3[i];
  }
  return r;
}

__global__ __launch_bounds__(32 * WAVES)
void attn_fwd_kernel(const float* __restrict__ q,
                     const float* __restrict__ k,
                     const float* __restrict__ v,
                     const float* __restrict__ scale_p,
                     const unsigned char* __restrict__ mask,
                     float* __restrict__ ctx_out,
                     float* __restrict__ attn_out) {
  __shared__ __align__(16) _Float16 lds_p[WAVES][16 * PSTRIDE];   // P tile (f16)
  __shared__ __align__(16) _Float16 lds_v[WAVES][64 * PSTRIDE];   // V^T chunk (f16)

  const int tid  = threadIdx.x;
  const int wave = tid >> 5;
  const int lane = tid & 31;
  const int hi   = lane >> 4;   // half-wave select
  const int ln   = lane & 15;

  const int gwave = blockIdx.x * WAVES + wave;   // 0 .. B*H*S/16-1
  const int bh    = gwave >> 7;                  // / (S/16)
  const int qb    = gwave & (NTILES - 1);
  const int qbase = qb * 16;

  const float scale = scale_p[0];

  const float* qrow  = q + ((size_t)bh * S_ + qbase + ln) * D_;
  const float* kbp   = k + (size_t)bh * S_ * D_;
  const float* vbp   = v + (size_t)bh * S_ * D_;
  const unsigned char* mrow = mask + ((size_t)bh * S_ + qbase) * S_;

  const v16h qa0 = load_qa(qrow, 0, hi);
  const v16h qa1 = load_qa(qrow, 1, hi);

  // ---------------- Pass 1: softmax denominators (online, no score storage) --
  float lsum[8];
#pragma unroll
  for (int r = 0; r < 8; ++r) lsum[r] = 0.0f;

  for (int t = 0; t < NTILES; ++t) {
    const float* krow = kbp + (size_t)(t * 16 + ln) * D_ + 16 * hi;
    v16h kb0 = load_kb(krow);
    v16h kb1 = load_kb(krow + 32);
    v8f acc = {};
    acc = wmma_f16(qa0, kb0, acc);
    acc = wmma_f16(qa1, kb1, acc);
    const int col = t * 16 + ln;
#pragma unroll
    for (int r = 0; r < 8; ++r) {
      float s = acc[r] * scale;
      unsigned char mb = mrow[(size_t)(8 * hi + r) * S_ + col];
      s = mb ? 1e-9f : s;                 // faithful masked_fill(1e-9)
      lsum[r] += __expf(fminf(s, 60.0f)); // clamp: overflow guard only
    }
  }
  // butterfly-reduce across the 16 lanes of each half-wave -> 1/rowsum
  float invl[8];
#pragma unroll
  for (int r = 0; r < 8; ++r) {
    float x = lsum[r];
    x += __shfl_xor(x, 1, 32);
    x += __shfl_xor(x, 2, 32);
    x += __shfl_xor(x, 4, 32);
    x += __shfl_xor(x, 8, 32);
    invl[r] = 1.0f / x;
  }

  // ---------------- Pass 2: attention out + P@V ----------------------------
  _Float16* pbuf = lds_p[wave];
  _Float16* vtb  = lds_v[wave];
  float* attn_row = attn_out + ((size_t)bh * S_ + qbase) * S_;

  v8f cacc[4] = {};   // 16x64 context accumulator (4 N-tiles of 16)

  for (int ch = 0; ch < NCHUNK; ++ch) {
    // stage V[32 keys][64] -> LDS as f16, transposed: vtb[dim][key]
#pragma unroll
    for (int kk = 0; kk < 16; ++kk) {
      const int key = ch * 32 + 2 * kk + hi;
      v4f vv = *(const v4f*)(vbp + (size_t)key * D_ + 4 * ln);
#pragma unroll
      for (int j = 0; j < 4; ++j)
        vtb[(4 * ln + j) * PSTRIDE + (2 * kk + hi)] = (_Float16)vv[j];
    }

    // two 16-key score tiles: recompute, normalize, emit attention + P(f16)
#pragma unroll
    for (int tt = 0; tt < 2; ++tt) {
      const int t = ch * 2 + tt;
      const float* krow = kbp + (size_t)(t * 16 + ln) * D_ + 16 * hi;
      v16h kb0 = load_kb(krow);
      v16h kb1 = load_kb(krow + 32);
      v8f acc = {};
      acc = wmma_f16(qa0, kb0, acc);
      acc = wmma_f16(qa1, kb1, acc);
      const int col = t * 16 + ln;
#pragma unroll
      for (int r = 0; r < 8; ++r) {
        float s = acc[r] * scale;
        unsigned char mb = mrow[(size_t)(8 * hi + r) * S_ + col];
        s = mb ? 1e-9f : s;
        float p = __expf(fminf(s, 60.0f)) * invl[r];
        attn_row[(size_t)(8 * hi + r) * S_ + col] = p;
        pbuf[(8 * hi + r) * PSTRIDE + tt * 16 + ln] = (_Float16)p;
      }
    }

    // same-wave LDS store->load ordering point
    asm volatile("s_wait_dscnt 0" ::: "memory");

    // A-fragment of P (16x32): row=ln, keys 16hi+0..15 (contiguous)
    v8h alo = *(const v8h*)(pbuf + ln * PSTRIDE + 16 * hi);
    v8h ahi = *(const v8h*)(pbuf + ln * PSTRIDE + 16 * hi + 8);
    v16h pa;
#pragma unroll
    for (int i = 0; i < 8; ++i) { pa[i] = alo[i]; pa[8 + i] = ahi[i]; }

#pragma unroll
    for (int nt = 0; nt < 4; ++nt) {
      // B-fragment of V (32x16): col n = nt*16+ln, 16 contiguous keys from 16hi
      v8h b0 = *(const v8h*)(vtb + (nt * 16 + ln) * PSTRIDE + 16 * hi);
      v8h b1 = *(const v8h*)(vtb + (nt * 16 + ln) * PSTRIDE + 16 * hi + 8);
      v16h vb;
#pragma unroll
      for (int i = 0; i < 8; ++i) { vb[i] = b0[i]; vb[8 + i] = b1[i]; }
      cacc[nt] = wmma_f16(pa, vb, cacc[nt]);
    }
  }

  // write context (16 rows x 64 dims), P already normalized
  float* crow = ctx_out + ((size_t)bh * S_ + qbase) * D_;
#pragma unroll
  for (int nt = 0; nt < 4; ++nt)
#pragma unroll
    for (int r = 0; r < 8; ++r)
      crow[(size_t)(8 * hi + r) * D_ + nt * 16 + ln] = cacc[nt][r];
}

extern "C" void kernel_launch(void* const* d_in, const int* in_sizes, int n_in,
                              void* d_out, int out_size, void* d_ws, size_t ws_size,
                              hipStream_t stream) {
  (void)in_sizes; (void)n_in; (void)out_size; (void)d_ws; (void)ws_size;
  const float* q = (const float*)d_in[0];
  const float* k = (const float*)d_in[1];
  const float* v = (const float*)d_in[2];
  const float* scale = (const float*)d_in[3];
  const unsigned char* mask = (const unsigned char*)d_in[4];

  float* ctx  = (float*)d_out;                              // [B,H,S,D]
  float* attn = ctx + (size_t)B_ * H_ * S_ * D_;            // [B,H,S,S]

  const int total_waves = B_ * H_ * (S_ / 16);              // 8192
  dim3 grid(total_waves / WAVES);                           // 1024
  dim3 block(32 * WAVES);                                   // 256
  hipLaunchKernelGGL(attn_fwd_kernel, grid, block, 0, stream,
                     q, k, v, scale, mask, ctx, attn);
}
